// MessagePassingLayer_49228915146779
// MI455X (gfx1250) — matrix operand order repaired and checked by reading, not compiled, not run on hardware
//
#include <hip/hip_runtime.h>
#include <stdint.h>

// ---------------------------------------------------------------------------
// MessagePassingLayer for MI455X (gfx1250), wave32 + WMMA bf16 path.
// d = 128. Edge MLP (257->128 GELU ->128) per edge, scatter-add, node MLP
// (256->128 GELU ->128), residual + LayerNorm.
// ---------------------------------------------------------------------------

typedef __attribute__((ext_vector_type(16))) __bf16 v16bf;
typedef __attribute__((ext_vector_type(8)))  float  v8f;

union FragBF { uint32_t u[8]; v16bf v; };

#define HID 128
#define WPB 4   // waves per block

__device__ __forceinline__ void lds_fence() {
    asm volatile("s_wait_dscnt 0" ::: "memory");
}

__device__ __forceinline__ void atomic_add_f32g(float* p, float v) {
    // native fp32 global atomic add (no CAS loop)
    asm volatile("global_atomic_add_f32 %0, %1, off"
                 :: "v"((unsigned long long)(uintptr_t)p), "v"(v) : "memory");
}

__device__ __forceinline__ unsigned short f2bf(float f) {
    uint32_t u = __float_as_uint(f);
    u += 0x7FFFu + ((u >> 16) & 1u);          // round-to-nearest-even
    return (unsigned short)(u >> 16);
}
__device__ __forceinline__ uint32_t pack2(float lo, float hi) {
    return (uint32_t)f2bf(lo) | ((uint32_t)f2bf(hi) << 16);
}
__device__ __forceinline__ float gelu_exact(float x) {
    return 0.5f * x * (1.0f + erff(x * 0.70710678118654752f));
}

// ---------------------------------------------------------------------------
// Pre-pack a row-major fp32 weight [K x 128] into per-lane WMMA B fragments:
// P[kt][nt][lane][r] = bf16 pair (W[k0][n], W[k0+1][n]),
//   n = nt*16 + lane%16,  k0 = kt*32 + (lane/16)*16 + 2r.
// Each lane's fragment = 8 contiguous dwords.
// ---------------------------------------------------------------------------
__global__ void prepack_b(const float* __restrict__ W, uint32_t* __restrict__ P,
                          int ktiles) {
    int i = blockIdx.x * blockDim.x + threadIdx.x;
    int total = ktiles * 8 * 32 * 8;
    if (i >= total) return;
    int r    =  i        & 7;
    int lane = (i >> 3)  & 31;
    int nt   = (i >> 8)  & 7;
    int kt   =  i >> 11;
    int n    = nt * 16 + (lane & 15);
    int k0   = kt * 32 + (lane >> 4) * 16 + 2 * r;
    P[i] = pack2(W[(size_t)k0 * HID + n], W[(size_t)(k0 + 1) * HID + n]);
}

__global__ void zero_f32(float* __restrict__ p, long long n) {
    long long i = (long long)blockIdx.x * blockDim.x + threadIdx.x;
    if (i < n) p[i] = 0.0f;
}

// A-fragment pair index: p(r,half) = half*8 + r
__device__ __forceinline__ int apair(int r, int half) {
    return (half << 3) + r;
}

// ---------------------------------------------------------------------------
// Edge kernel: one wave per 16-edge tile.
// ---------------------------------------------------------------------------
__global__ __launch_bounds__(WPB * 32)
void edge_kernel(const float* __restrict__ hidden,
                 const int*   __restrict__ eidx,     // [2, E]
                 const float* __restrict__ ew,
                 const float* __restrict__ We1,      // fp32, need row 256
                 const float* __restrict__ be1,
                 const float* __restrict__ be2,
                 const uint32_t* __restrict__ pWe1,  // packed [8][8][32][8]
                 const uint32_t* __restrict__ pWe2,  // packed [4][8][32][8]
                 float* __restrict__ agg,
                 int E, int etiles) {
    // per-wave LDS: A pairs [16][128] (8KB) + H pairs [16][64] (4KB) + ew/dst
    __shared__ uint32_t lds[WPB * (16 * 128 + 16 * 64 + 32)];
    const int wave = threadIdx.x >> 5;
    const int lane = threadIdx.x & 31;
    uint32_t* A   = lds + wave * (16 * 128 + 16 * 64 + 32);
    uint32_t* H   = A + 16 * 128;
    float*    EWs = (float*)(H + 16 * 64);
    int*      DST = (int*)(EWs + 16);

    const int tile = blockIdx.x * WPB + wave;
    if (tile >= etiles) return;                 // wave-uniform exit
    const int e0 = tile * 16;
    const int* src = eidx;
    const int* dst = eidx + E;

    const int nloc = lane & 15;
    const int half = lane >> 4;

    if (lane < 16) {
        int e = min(e0 + lane, E - 1);
        EWs[lane] = ew[e];
        DST[lane] = dst[e];
    }
    // gather + bf16-pack 16 edges x (src|dst) features
#pragma unroll 4
    for (int m = 0; m < 16; ++m) {
        int e = min(e0 + m, E - 1);
        int s = src[e];
        int d = dst[e];
        const float4 fs = *(const float4*)(hidden + (size_t)s * HID + lane * 4);
        const float4 fd = *(const float4*)(hidden + (size_t)d * HID + lane * 4);
        A[m * 128 +      lane * 2 + 0] = pack2(fs.x, fs.y);
        A[m * 128 +      lane * 2 + 1] = pack2(fs.z, fs.w);
        A[m * 128 + 64 + lane * 2 + 0] = pack2(fd.x, fd.y);
        A[m * 128 + 64 + lane * 2 + 1] = pack2(fd.z, fd.w);
    }
    lds_fence();

    // ---- GEMM1: [16x256] @ We1[0:256] + rank-1(ew x We1[256]) + be1, GELU ---
    unsigned short* H16 = (unsigned short*)H;
#pragma unroll
    for (int nt = 0; nt < 8; ++nt) {
        const int n = nt * 16 + nloc;
        const float b1   = be1[n];
        const float w256 = We1[(size_t)256 * HID + n];
        v8f acc;
#pragma unroll
        for (int r = 0; r < 8; ++r)
            acc[r] = b1 + EWs[r + 8 * half] * w256;
#pragma unroll
        for (int kt = 0; kt < 8; ++kt) {
            FragBF a, b;
#pragma unroll
            for (int r = 0; r < 8; ++r)
                a.u[r] = A[nloc * 128 + kt * 16 + apair(r, half)];
            const uint32_t* bp = pWe1 + (((size_t)kt * 8 + nt) * 32 + lane) * 8;
#pragma unroll
            for (int r = 0; r < 8; ++r) b.u[r] = bp[r];
            acc = __builtin_amdgcn_wmma_f32_16x16x32_bf16(
                false, a.v, false, b.v, (short)0, acc, false, false);
        }
#pragma unroll
        for (int r = 0; r < 8; ++r)
            H16[(r + 8 * half) * 128 + n] = f2bf(gelu_exact(acc[r]));
    }
    lds_fence();

    // ---- GEMM2: [16x128] @ We2 + be2 -> scatter-add to agg[dst] ----
#pragma unroll
    for (int nt = 0; nt < 8; ++nt) {
        const int n = nt * 16 + nloc;
        const float b2 = be2[n];
        v8f acc;
#pragma unroll
        for (int r = 0; r < 8; ++r) acc[r] = b2;
#pragma unroll
        for (int kt = 0; kt < 4; ++kt) {
            FragBF a, b;
#pragma unroll
            for (int r = 0; r < 8; ++r)
                a.u[r] = H[nloc * 64 + kt * 16 + apair(r, half)];
            const uint32_t* bp = pWe2 + (((size_t)kt * 8 + nt) * 32 + lane) * 8;
#pragma unroll
            for (int r = 0; r < 8; ++r) b.u[r] = bp[r];
            acc = __builtin_amdgcn_wmma_f32_16x16x32_bf16(
                false, a.v, false, b.v, (short)0, acc, false, false);
        }
#pragma unroll
        for (int r = 0; r < 8; ++r) {
            int m = r + 8 * half;
            atomic_add_f32g(&agg[(size_t)DST[m] * HID + n], acc[r]);
        }
    }
}

// ---------------------------------------------------------------------------
// Node kernel: one wave per 16-node tile. update MLP + residual + LayerNorm.
// ---------------------------------------------------------------------------
__global__ __launch_bounds__(WPB * 32)
void node_kernel(const float* __restrict__ hidden,
                 const float* __restrict__ agg,
                 const float* __restrict__ bu1,
                 const float* __restrict__ bu2,
                 const float* __restrict__ gamma,
                 const float* __restrict__ beta,
                 const uint32_t* __restrict__ pWu1,  // [8][8][32][8]
                 const uint32_t* __restrict__ pWu2,  // [4][8][32][8]
                 float* __restrict__ out,
                 int N, int ntiles) {
    __shared__ uint32_t lds[WPB * (16 * 128 + 16 * 64)];
    const int wave = threadIdx.x >> 5;
    const int lane = threadIdx.x & 31;
    uint32_t* A = lds + wave * (16 * 128 + 16 * 64);
    uint32_t* H = A + 16 * 128;
    float*    X = (float*)A;                    // overlays A after GEMM1

    const int tile = blockIdx.x * WPB + wave;
    if (tile >= ntiles) return;                 // wave-uniform exit
    const int n0 = tile * 16;
    const int nloc = lane & 15;
    const int half = lane >> 4;

#pragma unroll 4
    for (int m = 0; m < 16; ++m) {
        int node = min(n0 + m, N - 1);
        const float4 fh = *(const float4*)(hidden + (size_t)node * HID + lane * 4);
        const float4 fa = *(const float4*)(agg    + (size_t)node * HID + lane * 4);
        A[m * 128 +      lane * 2 + 0] = pack2(fh.x, fh.y);
        A[m * 128 +      lane * 2 + 1] = pack2(fh.z, fh.w);
        A[m * 128 + 64 + lane * 2 + 0] = pack2(fa.x, fa.y);
        A[m * 128 + 64 + lane * 2 + 1] = pack2(fa.z, fa.w);
    }
    lds_fence();

    // ---- GEMM1: [16x256] @ Wu1 + bu1, GELU ----
    unsigned short* H16 = (unsigned short*)H;
#pragma unroll
    for (int nt = 0; nt < 8; ++nt) {
        const int n = nt * 16 + nloc;
        const float b1 = bu1[n];
        v8f acc;
#pragma unroll
        for (int r = 0; r < 8; ++r) acc[r] = b1;
#pragma unroll
        for (int kt = 0; kt < 8; ++kt) {
            FragBF a, b;
#pragma unroll
            for (int r = 0; r < 8; ++r)
                a.u[r] = A[nloc * 128 + kt * 16 + apair(r, half)];
            const uint32_t* bp = pWu1 + (((size_t)kt * 8 + nt) * 32 + lane) * 8;
#pragma unroll
            for (int r = 0; r < 8; ++r) b.u[r] = bp[r];
            acc = __builtin_amdgcn_wmma_f32_16x16x32_bf16(
                false, a.v, false, b.v, (short)0, acc, false, false);
        }
#pragma unroll
        for (int r = 0; r < 8; ++r)
            H16[(r + 8 * half) * 128 + n] = f2bf(gelu_exact(acc[r]));
    }
    lds_fence();

    // ---- GEMM2: [16x128] @ Wu2 + bu2, residual into X ----
#pragma unroll
    for (int nt = 0; nt < 8; ++nt) {
        const int n = nt * 16 + nloc;
        const float b2 = bu2[n];
        v8f acc;
#pragma unroll
        for (int r = 0; r < 8; ++r) acc[r] = b2;
#pragma unroll
        for (int kt = 0; kt < 4; ++kt) {
            FragBF a, b;
#pragma unroll
            for (int r = 0; r < 8; ++r)
                a.u[r] = H[nloc * 64 + kt * 16 + apair(r, half)];
            const uint32_t* bp = pWu2 + (((size_t)kt * 8 + nt) * 32 + lane) * 8;
#pragma unroll
            for (int r = 0; r < 8; ++r) b.u[r] = bp[r];
            acc = __builtin_amdgcn_wmma_f32_16x16x32_bf16(
                false, a.v, false, b.v, (short)0, acc, false, false);
        }
#pragma unroll
        for (int r = 0; r < 8; ++r) {
            int m = r + 8 * half;
            int node = min(n0 + m, N - 1);
            float hv = hidden[(size_t)node * HID + n];
            X[m * 128 + n] = hv + acc[r];
        }
    }
    lds_fence();

    // ---- LayerNorm: lanes 0..15 each own one row ----
    if (lane < 16) {
        const int row  = lane;
        const int node = min(n0 + row, N - 1);
        float mu = 0.0f;
#pragma unroll 8
        for (int c = 0; c < HID; ++c) mu += X[row * 128 + c];
        mu *= (1.0f / HID);
        float var = 0.0f;
#pragma unroll 8
        for (int c = 0; c < HID; ++c) {
            float d = X[row * 128 + c] - mu;
            var += d * d;
        }
        var *= (1.0f / HID);
        const float rs = rsqrtf(var + 1e-5f);
#pragma unroll 8
        for (int c = 0; c < HID; ++c) {
            float v = (X[row * 128 + c] - mu) * rs * gamma[c] + beta[c];
            out[(size_t)node * HID + c] = v;
        }
    }
}

// ---------------------------------------------------------------------------
extern "C" void kernel_launch(void* const* d_in, const int* in_sizes, int n_in,
                              void* d_out, int out_size, void* d_ws, size_t ws_size,
                              hipStream_t stream) {
    const float* hidden = (const float*)d_in[0];
    const int*   eidx   = (const int*)  d_in[1];
    const float* ew     = (const float*)d_in[2];
    const float* We1    = (const float*)d_in[3];
    const float* be1    = (const float*)d_in[4];
    const float* We2    = (const float*)d_in[5];
    const float* be2    = (const float*)d_in[6];
    const float* Wu1    = (const float*)d_in[7];
    const float* bu1    = (const float*)d_in[8];
    const float* Wu2    = (const float*)d_in[9];
    const float* bu2    = (const float*)d_in[10];
    const float* gamma  = (const float*)d_in[11];
    const float* beta   = (const float*)d_in[12];
    float* out = (float*)d_out;

    const int N = in_sizes[0] / HID;
    const int E = in_sizes[2];

    // workspace layout
    char* ws = (char*)d_ws;
    float*    agg  = (float*)ws;                       // N*128 floats
    size_t off = (size_t)N * HID * sizeof(float);
    uint32_t* pWe1 = (uint32_t*)(ws + off); off += 8 * 8 * 32 * 8 * 4;
    uint32_t* pWe2 = (uint32_t*)(ws + off); off += 4 * 8 * 32 * 8 * 4;
    uint32_t* pWu1 = (uint32_t*)(ws + off); off += 8 * 8 * 32 * 8 * 4;
    uint32_t* pWu2 = (uint32_t*)(ws + off); off += 4 * 8 * 32 * 8 * 4;

    // 1) zero aggregation buffer (every launch; harness doesn't re-poison)
    {
        long long n = (long long)N * HID;
        int blk = 256;
        int grd = (int)((n + blk - 1) / blk);
        hipLaunchKernelGGL(zero_f32, dim3(grd), dim3(blk), 0, stream, agg, n);
    }
    // 2) pre-pack weights into WMMA B-fragment layout
    {
        int blk = 256;
        hipLaunchKernelGGL(prepack_b, dim3((8 * 2048 + blk - 1) / blk), dim3(blk), 0, stream, We1, pWe1, 8);
        hipLaunchKernelGGL(prepack_b, dim3((4 * 2048 + blk - 1) / blk), dim3(blk), 0, stream, We2, pWe2, 4);
        hipLaunchKernelGGL(prepack_b, dim3((8 * 2048 + blk - 1) / blk), dim3(blk), 0, stream, Wu1, pWu1, 8);
        hipLaunchKernelGGL(prepack_b, dim3((4 * 2048 + blk - 1) / blk), dim3(blk), 0, stream, Wu2, pWu2, 4);
    }
    // 3) edge MLP + scatter
    {
        int etiles = (E + 15) / 16;
        int blocks = (etiles + WPB - 1) / WPB;
        hipLaunchKernelGGL(edge_kernel, dim3(blocks), dim3(WPB * 32), 0, stream,
                           hidden, eidx, ew, We1, be1, be2, pWe1, pWe2, agg, E, etiles);
    }
    // 4) node MLP + residual + LayerNorm
    {
        int ntiles = (N + 15) / 16;
        int blocks = (ntiles + WPB - 1) / WPB;
        hipLaunchKernelGGL(node_kernel, dim3(blocks), dim3(WPB * 32), 0, stream,
                           hidden, agg, bu1, bu2, gamma, beta, pWu1, pWu2, out, N, ntiles);
    }
}
